// Seq2SeqWTopic_71837622993507
// MI455X (gfx1250) — matrix operand order repaired and checked by reading, not compiled, not run on hardware
//
#include <hip/hip_runtime.h>
#include <math.h>

// ---------------- model dims ----------------
#define LSTM_ 128
#define H2_   256
#define E_    1024
#define V_    30611
#define A_    64
#define B_    16
#define S_    400
#define T_    96
#define NT_   1914          // ceil(V/16)
#define VPAD_ (NT_*16)      // 30624
#define START_TOK_ 101

typedef __attribute__((ext_vector_type(16))) __bf16 v16bf;
typedef __attribute__((ext_vector_type(8)))  float  v8f;

#define WMMA_BF16(a,b,c) \
  __builtin_amdgcn_wmma_f32_16x16x32_bf16(false,(a),false,(b),(short)0,(c),false,false)

__device__ __forceinline__ unsigned short f2bf(float f) {
  unsigned u = __float_as_uint(f);
  unsigned r = (u + 0x7FFFu + ((u >> 16) & 1u)) >> 16;
  return (unsigned short)r;
}
__device__ __forceinline__ float sigmoidf_(float x) { return 1.0f / (1.0f + __expf(-x)); }

// A fragment: 16x32 bf16, row-major source, lane m=lane&15 owns row m.
// elems e -> K = hi*16 + e
__device__ __forceinline__ v16bf load_fragA(const unsigned short* p, int ld, int lane) {
  int m = lane & 15, hi = lane >> 4;
  const unsigned* q = (const unsigned*)(p + m * ld + hi * 16);
  union { v16bf v; unsigned u[8]; } r;
#pragma unroll
  for (int i = 0; i < 8; ++i) r.u[i] = q[i];
  return r.v;
}
// B fragment: 32x16 bf16 (K x N), source is weight row-major [N][K].
// lane n=lane&15 owns column n; elems e -> K = hi*16 + e
__device__ __forceinline__ v16bf load_fragB(const unsigned short* p, int ld, int lane) {
  int n = lane & 15, hi = lane >> 4;
  const unsigned* q = (const unsigned*)(p + n * ld + hi * 16);
  union { v16bf v; unsigned u[8]; } r;
#pragma unroll
  for (int i = 0; i < 8; ++i) r.u[i] = q[i];
  return r.v;
}

// ---------------- elementwise / prep kernels ----------------
__global__ void k_f32_to_bf16(const float* src, unsigned short* dst, int n_src, int n_tot) {
  int i = blockIdx.x * 256 + threadIdx.x;
  if (i < n_tot) dst[i] = (i < n_src) ? f2bf(src[i]) : (unsigned short)0;
}

// WihE_bf [1024][1024] = dec_Wih[:, :E] ; Wcomb_bf [1024][512] = [dec_Wih[:,E:] | dec_Whh]
__global__ void k_prep_dec_w(const float* dec_Wih, const float* dec_Whh,
                             unsigned short* WihE, unsigned short* Wcomb) {
  int i = blockIdx.x * 256 + threadIdx.x;
  if (i >= 1024 * 1536) return;
  int n = i / 1536, j = i % 1536;
  if (j < 1024) {
    WihE[n * 1024 + j] = f2bf(dec_Wih[n * 1280 + j]);
  } else {
    int k = j - 1024; // 0..511
    float v = (k < 256) ? dec_Wih[n * 1280 + 1024 + k] : dec_Whh[n * 256 + (k - 256)];
    Wcomb[n * 512 + k] = f2bf(v);
  }
}

__global__ void k_embed_enc(const float* emb, const int* article, unsigned short* out) {
  int i = blockIdx.x * 256 + threadIdx.x;
  if (i >= S_ * B_ * E_) return;
  int row = i >> 10, e = i & 1023;       // row = s*16 + b
  int s = row >> 4, b = row & 15;
  int tok = article[b * S_ + s];
  out[i] = f2bf(emb[(long long)tok * E_ + e]);
}

__global__ void k_embed_dec(const float* emb, const int* abstract, unsigned short* out) {
  int i = blockIdx.x * 256 + threadIdx.x;
  if (i >= T_ * B_ * E_) return;
  int row = i >> 10, e = i & 1023;       // row = t*16 + b
  int t = row >> 4, b = row & 15;
  int tok = (t == 0) ? START_TOK_ : abstract[b * T_ + (t - 1)];
  out[i] = f2bf(emb[(long long)tok * E_ + e]);
}

// ---------------- generic WMMA GEMM: out[M,N] = A[M,K] @ W[N,K]^T + bias ----------------
__global__ void k_gemm_bf16(const unsigned short* __restrict__ A,
                            const unsigned short* __restrict__ W,
                            const float* __restrict__ bias,
                            float* __restrict__ out, int M, int N, int K) {
  int wave = threadIdx.x >> 5, lane = threadIdx.x & 31;
  int row0 = blockIdx.x * 16;
  int col0 = (blockIdx.y * 8 + wave) * 16;
  if (col0 >= N) return;
  v8f acc = {};
  for (int k0 = 0; k0 < K; k0 += 32) {
    v16bf a = load_fragA(A + (long long)row0 * K + k0, K, lane);
    v16bf b = load_fragB(W + (long long)col0 * K + k0, K, lane);
    acc = WMMA_BF16(a, b, acc);
  }
  int n = col0 + (lane & 15), hi = lane >> 4;
  float bv = bias[n];
#pragma unroll
  for (int r = 0; r < 8; ++r) {
    int m = row0 + r + hi * 8;
    out[(long long)m * N + n] = acc[r] + bv;
  }
}

// ---------------- encoder recurrence (1 workgroup per direction, 16 waves) ----------------
__global__ void k_enc_recurrent(const float* __restrict__ pre_f, const float* __restrict__ pre_b,
                                const unsigned short* __restrict__ Whh_f,
                                const unsigned short* __restrict__ Whh_b,
                                float* __restrict__ enc_out, unsigned short* __restrict__ enc_out_bf,
                                float* __restrict__ h0, float* __restrict__ c0) {
  int dir = blockIdx.x;
  const float* pre = dir ? pre_b : pre_f;
  const unsigned short* Whh = dir ? Whh_b : Whh_f;
  __shared__ __attribute__((aligned(16))) unsigned short hbf[16 * 128];
  __shared__ float cst[16 * 128];
  __shared__ float g[16 * 512];
  int tid = threadIdx.x, wave = tid >> 5, lane = tid & 31;
  for (int i = tid; i < 16 * 128; i += 512) { hbf[i] = 0; cst[i] = 0.0f; }
  __syncthreads();
  for (int step = 0; step < S_; ++step) {
    int s = dir ? (S_ - 1 - step) : step;
    // g = h @ Whh^T + pre  (N=512 -> 2 tiles per wave, K=128)
#pragma unroll
    for (int tt = 0; tt < 2; ++tt) {
      int col0 = (wave * 2 + tt) * 16;
      v8f acc = {};
#pragma unroll
      for (int k0 = 0; k0 < 128; k0 += 32) {
        v16bf a = load_fragA((const unsigned short*)hbf + k0, 128, lane);
        v16bf b = load_fragB(Whh + col0 * 128 + k0, 128, lane);
        acc = WMMA_BF16(a, b, acc);
      }
      int n = col0 + (lane & 15), hi = lane >> 4;
#pragma unroll
      for (int r = 0; r < 8; ++r) {
        int bb = r + hi * 8;
        g[bb * 512 + n] = acc[r] + pre[((long long)s * 16 + bb) * 512 + n];
      }
    }
    __syncthreads();
    // gates (i,f,g,o)
    for (int p = tid; p < 16 * 128; p += 512) {
      int bb = p >> 7, j = p & 127;
      float gi = g[bb * 512 + j],       gf = g[bb * 512 + 128 + j];
      float gg = g[bb * 512 + 256 + j], go = g[bb * 512 + 384 + j];
      float cn = sigmoidf_(gf) * cst[p] + sigmoidf_(gi) * tanhf(gg);
      float h  = sigmoidf_(go) * tanhf(cn);
      cst[p] = cn;
      hbf[p] = f2bf(h);
      long long eo = ((long long)bb * S_ + s) * 256 + dir * 128 + j;
      enc_out[eo] = h;
      enc_out_bf[eo] = f2bf(h);
      if (step == S_ - 1) { h0[bb * 256 + dir * 128 + j] = h; c0[bb * 256 + dir * 128 + j] = cn; }
    }
    __syncthreads();
  }
}

// ---------------- decoder recurrence + attention (1 workgroup, 16 waves) ----------------
__global__ void k_decoder(const float* __restrict__ pre_dec,
                          const unsigned short* __restrict__ Wcomb,
                          const unsigned short* __restrict__ attn_hW_bf,
                          const float* __restrict__ attn_h_b,
                          const float* __restrict__ attn_fW,
                          const float* __restrict__ attn_f_b,
                          const float* __restrict__ enc_ws,
                          const float* __restrict__ enc_out,
                          const int* __restrict__ art_len,
                          const float* __restrict__ h0, const float* __restrict__ c0,
                          unsigned short* __restrict__ hbf_out) {
  __shared__ __attribute__((aligned(16))) unsigned short xcat[16 * 512]; // [b][ctx(256)|h(256)]
  __shared__ float cst[16 * 256];
  __shared__ float g[16 * 1024];
  __shared__ float wh[16 * 64];
  __shared__ float wsc[16 * 400];
  __shared__ float fw[64];
  int tid = threadIdx.x, wave = tid >> 5, lane = tid & 31;
  for (int p = tid; p < 16 * 256; p += 512) {
    int b = p >> 8, j = p & 255;
    xcat[b * 512 + j] = 0;                  // ctx = 0
    xcat[b * 512 + 256 + j] = f2bf(h0[p]);  // h
    cst[p] = c0[p];
  }
  if (tid < 64) fw[tid] = attn_fW[tid];
  __syncthreads();
  for (int t = 0; t < T_; ++t) {
    // g = [ctx|h] @ Wcomb^T + pre_dec   (N=1024 -> 4 tiles/wave, K=512)
#pragma unroll
    for (int tt = 0; tt < 4; ++tt) {
      int col0 = (wave * 4 + tt) * 16;
      v8f acc = {};
      for (int k0 = 0; k0 < 512; k0 += 32) {
        v16bf a = load_fragA((const unsigned short*)xcat + k0, 512, lane);
        v16bf b = load_fragB(Wcomb + col0 * 512 + k0, 512, lane);
        acc = WMMA_BF16(a, b, acc);
      }
      int n = col0 + (lane & 15), hi = lane >> 4;
#pragma unroll
      for (int r = 0; r < 8; ++r) {
        int bb = r + hi * 8;
        g[bb * 1024 + n] = acc[r] + pre_dec[((long long)t * 16 + bb) * 1024 + n];
      }
    }
    __syncthreads();
    // gates (H2=256)
    for (int p = tid; p < 16 * 256; p += 512) {
      int b = p >> 8, j = p & 255;
      float gi = g[b * 1024 + j],       gf = g[b * 1024 + 256 + j];
      float gg = g[b * 1024 + 512 + j], go = g[b * 1024 + 768 + j];
      float cn = sigmoidf_(gf) * cst[p] + sigmoidf_(gi) * tanhf(gg);
      float h  = sigmoidf_(go) * tanhf(cn);
      cst[p] = cn;
      unsigned short hb = f2bf(h);
      xcat[b * 512 + 256 + j] = hb;
      hbf_out[((long long)t * 16 + b) * 256 + j] = hb;
    }
    __syncthreads();
    // wh = h @ attn_h_W^T + b   (waves 0..3, M=16 N=64 K=256)
    if (wave < 4) {
      int col0 = wave * 16;
      v8f acc = {};
#pragma unroll
      for (int k0 = 0; k0 < 256; k0 += 32) {
        v16bf a = load_fragA((const unsigned short*)xcat + 256 + k0, 512, lane);
        v16bf b = load_fragB(attn_hW_bf + col0 * 256 + k0, 256, lane);
        acc = WMMA_BF16(a, b, acc);
      }
      int n = col0 + (lane & 15), hi = lane >> 4;
#pragma unroll
      for (int r = 0; r < 8; ++r) wh[(r + hi * 8) * 64 + n] = acc[r] + attn_h_b[n];
    }
    __syncthreads();
    // scores: tanh(enc_ws + wh) . fW + fb, masked
    for (int p = tid; p < 16 * 400; p += 512) {
      int b = p / 400, s = p % 400;
      const float* ew = enc_ws + ((long long)b * S_ + s) * 64;
      float sc = attn_f_b[0];
#pragma unroll 8
      for (int a2 = 0; a2 < 64; ++a2) sc += tanhf(ew[a2] + wh[b * 64 + a2]) * fw[a2];
      int len = art_len[b]; len = len < 1 ? 1 : (len > S_ ? S_ : len);
      float mk = (s < len) ? 9999.0f : -9999.0f;
      wsc[p] = fminf(sc, mk);
    }
    __syncthreads();
    // softmax over S per batch row: wave w handles b = w
    {
      int b = wave;
      float mx = -1e30f;
      for (int s = lane; s < S_; s += 32) mx = fmaxf(mx, wsc[b * 400 + s]);
      for (int o = 16; o; o >>= 1) mx = fmaxf(mx, __shfl_xor(mx, o, 32));
      float sum = 0.f;
      for (int s = lane; s < S_; s += 32) {
        float e = __expf(wsc[b * 400 + s] - mx);
        wsc[b * 400 + s] = e; sum += e;
      }
      for (int o = 16; o; o >>= 1) sum += __shfl_xor(sum, o, 32);
      float inv = 1.0f / sum;
      for (int s = lane; s < S_; s += 32) wsc[b * 400 + s] *= inv;
    }
    __syncthreads();
    // ctx = w @ enc_out  (per-b matvec)
    for (int p = tid; p < 16 * 256; p += 512) {
      int b = p >> 8, d = p & 255;
      const float* eo = enc_out + (long long)b * S_ * 256 + d;
      const float* wv = wsc + b * 400;
      float cv = 0.f;
      for (int s = 0; s < S_; ++s) cv += wv[s] * eo[s * 256];
      xcat[b * 512 + d] = f2bf(cv);
    }
    __syncthreads();
  }
}

// ---------------- fused logits GEMM + online logsumexp + NLL ----------------
__global__ void k_zero_acc(float* acc) { acc[0] = 0.f; acc[1] = 0.f; }

__global__ void k_logits_loss(const unsigned short* __restrict__ hbf,
                              const unsigned short* __restrict__ predW_bf,
                              const float* __restrict__ pred_b,
                              const int* __restrict__ abstract,
                              float* __restrict__ acc2) {
  __shared__ __attribute__((aligned(16))) unsigned short ash[16 * 256];
  __shared__ int tgt[16];
  __shared__ float mrg_m[8 * 16], mrg_s[8 * 16], mrg_t[8 * 16];
  __shared__ float redbuf[32];
  int tid = threadIdx.x, wave = tid >> 5, lane = tid & 31;
  int t = blockIdx.x;
  for (int i = tid; i < 16 * 256; i += 256) ash[i] = hbf[((long long)t * 16) * 256 + i];
  if (tid < 16) tgt[tid] = abstract[tid * T_ + t];
  __syncthreads();
  v16bf afrag[8];
#pragma unroll
  for (int kk = 0; kk < 8; ++kk)
    afrag[kk] = load_fragA((const unsigned short*)ash + kk * 32, 256, lane);
  int hi = lane >> 4, nl = lane & 15;
  float rm[8], rs[8], tl[8];
#pragma unroll
  for (int r = 0; r < 8; ++r) { rm[r] = -INFINITY; rs[r] = 0.f; tl[r] = -INFINITY; }
  for (int j = wave; j < NT_; j += 8) {
    int n0 = j * 16;
    v8f acc = {};
#pragma unroll
    for (int kk = 0; kk < 8; ++kk) {
      v16bf b = load_fragB(predW_bf + (long long)n0 * 256 + kk * 32, 256, lane);
      acc = WMMA_BF16(afrag[kk], b, acc);
    }
    int n = n0 + nl;
    bool nv = (n < V_);
    float bias = nv ? pred_b[n] : 0.f;
#pragma unroll
    for (int r = 0; r < 8; ++r) {
      int row = r + hi * 8;
      float v = nv ? (acc[r] + bias) : -INFINITY;
      float tmax = v;
      for (int o = 8; o; o >>= 1) tmax = fmaxf(tmax, __shfl_xor(tmax, o, 16));
      float pe = (v == -INFINITY) ? 0.f : __expf(v - tmax);
      float tsum = pe;
      for (int o = 8; o; o >>= 1) tsum += __shfl_xor(tsum, o, 16);
      float nm = fmaxf(rm[r], tmax);
      rs[r] = rs[r] * __expf(rm[r] - nm) + tsum * __expf(tmax - nm);
      rm[r] = nm;
      if (nv && n == tgt[row]) tl[r] = v;
    }
  }
#pragma unroll
  for (int r = 0; r < 8; ++r)
    for (int o = 8; o; o >>= 1) tl[r] = fmaxf(tl[r], __shfl_xor(tl[r], o, 16));
  if (nl == 0) {
#pragma unroll
    for (int r = 0; r < 8; ++r) {
      int row = r + hi * 8;
      mrg_m[wave * 16 + row] = rm[r];
      mrg_s[wave * 16 + row] = rs[r];
      mrg_t[wave * 16 + row] = tl[r];
    }
  }
  __syncthreads();
  if (tid < 16) {
    int row = tid;
    float M = -INFINITY;
    for (int w = 0; w < 8; ++w) M = fmaxf(M, mrg_m[w * 16 + row]);
    float Ssum = 0.f, TL = -INFINITY;
    for (int w = 0; w < 8; ++w) {
      Ssum += mrg_s[w * 16 + row] * __expf(mrg_m[w * 16 + row] - M);
      TL = fmaxf(TL, mrg_t[w * 16 + row]);
    }
    float nll = __logf(Ssum) + M - TL;
    float valid = (tgt[row] != 0) ? 1.f : 0.f;
    redbuf[row] = nll * valid;
    redbuf[16 + row] = valid;
  }
  __syncthreads();
  if (tid == 0) {
    float s1 = 0.f, s2 = 0.f;
    for (int i = 0; i < 16; ++i) { s1 += redbuf[i]; s2 += redbuf[16 + i]; }
    atomicAdd(&acc2[0], s1);
    atomicAdd(&acc2[1], s2);
  }
}

__global__ void k_finalize(const float* acc2, float* out) {
  out[0] = acc2[0] / fmaxf(acc2[1], 1.0f);
}

// ---------------- host orchestration ----------------
extern "C" void kernel_launch(void* const* d_in, const int* in_sizes, int n_in,
                              void* d_out, int out_size, void* d_ws, size_t ws_size,
                              hipStream_t stream) {
  const float* word_emb  = (const float*)d_in[0];
  const int*   article   = (const int*)d_in[1];
  const int*   art_len   = (const int*)d_in[2];
  const int*   abstract  = (const int*)d_in[3];
  const float* enc_Wih_f = (const float*)d_in[4];
  const float* enc_Whh_f = (const float*)d_in[5];
  const float* enc_b_f   = (const float*)d_in[6];
  const float* enc_Wih_b = (const float*)d_in[7];
  const float* enc_Whh_b = (const float*)d_in[8];
  const float* enc_b_b   = (const float*)d_in[9];
  const float* dec_Wih   = (const float*)d_in[10];
  const float* dec_Whh   = (const float*)d_in[11];
  const float* dec_b     = (const float*)d_in[12];
  const float* pred_W    = (const float*)d_in[13];
  const float* pred_b    = (const float*)d_in[14];
  const float* attn_s_W  = (const float*)d_in[15];
  const float* attn_s_b  = (const float*)d_in[16];
  const float* attn_h_W  = (const float*)d_in[17];
  const float* attn_h_b  = (const float*)d_in[18];
  const float* attn_f_W  = (const float*)d_in[19];
  const float* attn_f_b  = (const float*)d_in[20];

  char* ws = (char*)d_ws; size_t off = 0;
  auto alloc = [&](size_t bytes) -> void* {
    void* p = ws + off; off += (bytes + 255) & ~(size_t)255; return p;
  };
  unsigned short* Xe_bf      = (unsigned short*)alloc((size_t)S_ * B_ * E_ * 2);
  unsigned short* embD_bf    = (unsigned short*)alloc((size_t)T_ * B_ * E_ * 2);
  unsigned short* Wihf_bf    = (unsigned short*)alloc((size_t)512 * E_ * 2);
  unsigned short* Wihb_bf    = (unsigned short*)alloc((size_t)512 * E_ * 2);
  unsigned short* Whhf_bf    = (unsigned short*)alloc((size_t)512 * 128 * 2);
  unsigned short* Whhb_bf    = (unsigned short*)alloc((size_t)512 * 128 * 2);
  unsigned short* WihE_bf    = (unsigned short*)alloc((size_t)1024 * 1024 * 2);
  unsigned short* Wcomb_bf   = (unsigned short*)alloc((size_t)1024 * 512 * 2);
  unsigned short* attnS_bf   = (unsigned short*)alloc((size_t)64 * 256 * 2);
  unsigned short* attnH_bf   = (unsigned short*)alloc((size_t)64 * 256 * 2);
  unsigned short* predW_bf   = (unsigned short*)alloc((size_t)VPAD_ * 256 * 2);
  float*          pre_f      = (float*)alloc((size_t)S_ * B_ * 512 * 4);
  float*          pre_b      = (float*)alloc((size_t)S_ * B_ * 512 * 4);
  float*          enc_out    = (float*)alloc((size_t)B_ * S_ * 256 * 4);
  unsigned short* enc_out_bf = (unsigned short*)alloc((size_t)B_ * S_ * 256 * 2);
  float*          enc_ws     = (float*)alloc((size_t)B_ * S_ * 64 * 4);
  float*          h0buf      = (float*)alloc((size_t)B_ * 256 * 4);
  float*          c0buf      = (float*)alloc((size_t)B_ * 256 * 4);
  float*          pre_dec    = (float*)alloc((size_t)T_ * B_ * 1024 * 4);
  unsigned short* hbf_store  = (unsigned short*)alloc((size_t)T_ * B_ * 256 * 2);
  float*          acc2       = (float*)alloc(2 * 4);

  auto blk = [](long long n) { return (unsigned)((n + 255) / 256); };

  // weight conversions
  k_f32_to_bf16<<<blk(512 * E_), 256, 0, stream>>>(enc_Wih_f, Wihf_bf, 512 * E_, 512 * E_);
  k_f32_to_bf16<<<blk(512 * E_), 256, 0, stream>>>(enc_Wih_b, Wihb_bf, 512 * E_, 512 * E_);
  k_f32_to_bf16<<<blk(512 * 128), 256, 0, stream>>>(enc_Whh_f, Whhf_bf, 512 * 128, 512 * 128);
  k_f32_to_bf16<<<blk(512 * 128), 256, 0, stream>>>(enc_Whh_b, Whhb_bf, 512 * 128, 512 * 128);
  k_f32_to_bf16<<<blk(64 * 256), 256, 0, stream>>>(attn_s_W, attnS_bf, 64 * 256, 64 * 256);
  k_f32_to_bf16<<<blk(64 * 256), 256, 0, stream>>>(attn_h_W, attnH_bf, 64 * 256, 64 * 256);
  k_f32_to_bf16<<<blk((long long)VPAD_ * 256), 256, 0, stream>>>(
      pred_W, predW_bf, V_ * 256, VPAD_ * 256);
  k_prep_dec_w<<<blk(1024 * 1536), 256, 0, stream>>>(dec_Wih, dec_Whh, WihE_bf, Wcomb_bf);

  // embedding gathers
  k_embed_enc<<<blk((long long)S_ * B_ * E_), 256, 0, stream>>>(word_emb, article, Xe_bf);
  k_embed_dec<<<blk((long long)T_ * B_ * E_), 256, 0, stream>>>(word_emb, abstract, embD_bf);

  // encoder input GEMMs: [6400,1024] x [1024,512]
  k_gemm_bf16<<<dim3(S_ * B_ / 16, 4), 256, 0, stream>>>(Xe_bf, Wihf_bf, enc_b_f, pre_f,
                                                         S_ * B_, 512, E_);
  k_gemm_bf16<<<dim3(S_ * B_ / 16, 4), 256, 0, stream>>>(Xe_bf, Wihb_bf, enc_b_b, pre_b,
                                                         S_ * B_, 512, E_);
  // encoder recurrence
  k_enc_recurrent<<<2, 512, 0, stream>>>(pre_f, pre_b, Whhf_bf, Whhb_bf,
                                         enc_out, enc_out_bf, h0buf, c0buf);
  // enc_ws = enc_out @ attn_s_W^T + b : [6400,256] x [256,64]
  k_gemm_bf16<<<dim3(S_ * B_ / 16, 1), 256, 0, stream>>>(enc_out_bf, attnS_bf, attn_s_b,
                                                         enc_ws, S_ * B_, 64, 256);
  // decoder input GEMM: [1536,1024] x [1024,1024]
  k_gemm_bf16<<<dim3(T_ * B_ / 16, 8), 256, 0, stream>>>(embD_bf, WihE_bf, dec_b, pre_dec,
                                                         T_ * B_, 1024, E_);
  // decoder recurrence + attention
  k_decoder<<<1, 512, 0, stream>>>(pre_dec, Wcomb_bf, attnH_bf, attn_h_b, attn_f_W, attn_f_b,
                                   enc_ws, enc_out, art_len, h0buf, c0buf, hbf_store);
  // fused vocab GEMM + logsumexp + NLL
  k_zero_acc<<<1, 1, 0, stream>>>(acc2);
  k_logits_loss<<<T_, 256, 0, stream>>>(hbf_store, predW_bf, pred_b, abstract, acc2);
  k_finalize<<<1, 1, 0, stream>>>(acc2, (float*)d_out);
}
